// Generator_56736517980496
// MI455X (gfx1250) — matrix operand order repaired and checked by reading, not compiled, not run on hardware
//
#include <hip/hip_runtime.h>
#include <math.h>

typedef __attribute__((ext_vector_type(8)))  float  v8f;
typedef __attribute__((ext_vector_type(16))) __bf16 v16bf;

#define B_SZ   16384
#define E_SZ   256
#define H_SZ   512
#define F_SZ   514
#define V_SZ   1024
#define SEQ_L  8192
#define IN_SZ  1282   // 3*E + F

// Split fp32 into bf16 hi (truncation) + bf16 lo (RNE of exact residual).
// a ~= hi + lo with |err| <~ 2^-17 |a|;  a*w ~= hi*wh + hi*wl + lo*wh
// (dropped lo*wl term <= 2^-16 relative) -> fp32-like GEMM accuracy.
__device__ __forceinline__ void bf16_split(float f, __bf16& hi, __bf16& lo) {
  union { float f; unsigned u; } c; c.f = f;
  union { unsigned short s; __bf16 b; } hb; hb.s = (unsigned short)(c.u >> 16);
  hi = hb.b;
  union { unsigned u; float f; } hf; hf.u = c.u & 0xFFFF0000u;
  lo = (__bf16)(f - hf.f);   // exact residual, RNE to bf16
}

// ---------------------------------------------------------------------------
// Kernel 1: assemble x = [emb | pos_emb | ho_pos @ w_hp.T + b_hp | cond_data]
// ---------------------------------------------------------------------------
__global__ void assemble_x_kernel(const float* __restrict__ cond,
                                  const float* __restrict__ ho_pos,
                                  const float* __restrict__ emb_table,
                                  const float* __restrict__ pos_table,
                                  const float* __restrict__ w_hp,
                                  const float* __restrict__ b_hp,
                                  const int*   __restrict__ type_label,
                                  const int*   __restrict__ pos,
                                  float* __restrict__ x) {
  long long idx = (long long)blockIdx.x * blockDim.x + threadIdx.x;
  const long long total = (long long)B_SZ * IN_SZ;
  if (idx >= total) return;
  int b = (int)(idx / IN_SZ);
  int c = (int)(idx % IN_SZ);
  float v;
  if (c < E_SZ) {
    v = emb_table[(long long)type_label[b] * E_SZ + c];
  } else if (c < 2 * E_SZ) {
    int p = pos[b] % SEQ_L; if (p < 0) p += SEQ_L;
    v = pos_table[(long long)p * E_SZ + (c - E_SZ)];
  } else if (c < 3 * E_SZ) {
    int e = c - 2 * E_SZ;
    v = ho_pos[b * 2 + 0] * w_hp[e * 2 + 0] +
        ho_pos[b * 2 + 1] * w_hp[e * 2 + 1] + b_hp[e];
  } else {
    v = cond[(long long)b * F_SZ + (c - 3 * E_SZ)];
  }
  x[idx] = v;
}

// ---------------------------------------------------------------------------
// Kernel 2: C[M,N] = act(A[M,K] @ W[N,K]^T + bias)
// bf16x3 split-GEMM on v_wmma_f32_16x16x32_bf16 (f32 accumulate).
// block = 256 threads (8 wave32), C tile 128(M) x 64(N), K staged 32 at a time.
// act: 0 = identity, 1 = leaky_relu(0.01)
// Assumes M % 128 == 0 and N % 64 == 0 (true for all call sites here).
// ---------------------------------------------------------------------------
#define LDK 40   // row stride in halves: 80B, 16B-multiple, bank-spreading
__global__ __launch_bounds__(256)
void gemm_bias_kernel(const float* __restrict__ A, int lda,
                      const float* __restrict__ W, int ldw,
                      const float* __restrict__ bias,
                      float* __restrict__ C, int ldc,
                      int K, int act) {
  __shared__ __bf16 AsH[128][LDK];
  __shared__ __bf16 AsL[128][LDK];
  __shared__ __bf16 WsH[64][LDK];
  __shared__ __bf16 WsL[64][LDK];

  const int tid  = threadIdx.x;
  const int lane = tid & 31;
  const int wave = tid >> 5;
  const int wm   = wave & 3;      // wave's 32-row band within 128
  const int wn   = wave >> 2;     // wave's 32-col band within 64
  const int half = lane >> 4;     // 0: lanes 0-15, 1: lanes 16-31
  const int l15  = lane & 15;

  const int rowBase = blockIdx.y * 128;
  const int colBase = blockIdx.x * 64;

  v8f acc[2][2];
  acc[0][0] = (v8f){}; acc[0][1] = (v8f){};
  acc[1][0] = (v8f){}; acc[1][1] = (v8f){};

  // A fragment K-groups (ISA 16-bit A 16x32 layout):
  //  lanes 0-15: elems 0..7 <- K 0..7,  elems 8..15 <- K 16..23
  //  lanes16-31: elems 0..7 <- K 8..15, elems 8..15 <- K 24..31
  const int kA = half * 8;
  const int kB = 16 + half * 8;
  // B fragment: same K-groups as A (elems 0..7 <- K 8h..8h+7, elems 8..15 <-
  // K 16+8h..16+8h+7).
  const int kOff = half * 16;

  const int ktiles = (K + 31) >> 5;
  for (int kt = 0; kt < ktiles; ++kt) {
    const int k0 = kt << 5;
    // Stage A: 128x32 fp32 -> bf16 hi/lo (zero-fill K tail)
    #pragma unroll
    for (int i = 0; i < 16; ++i) {
      int l = tid + i * 256;
      int m = l >> 5, k = l & 31;
      float v = 0.0f;
      if (k0 + k < K) v = A[(long long)(rowBase + m) * lda + (k0 + k)];
      __bf16 h, lo; bf16_split(v, h, lo);
      AsH[m][k] = h; AsL[m][k] = lo;
    }
    // Stage W: 64(N) x 32(K) fp32 -> bf16 hi/lo, same [n][k] orientation
    #pragma unroll
    for (int i = 0; i < 8; ++i) {
      int l = tid + i * 256;
      int n = l >> 5, k = l & 31;
      float v = 0.0f;
      if (k0 + k < K) v = W[(long long)(colBase + n) * ldw + (k0 + k)];
      __bf16 h, lo; bf16_split(v, h, lo);
      WsH[n][k] = h; WsL[n][k] = lo;
    }
    // Prefetch next K tile into cache (global_prefetch_b8)
    if (k0 + 32 < K) {
      int m = tid >> 5, k = tid & 31;
      __builtin_prefetch(&A[(long long)(rowBase + m) * lda + (k0 + 32 + k)], 0, 3);
      if (tid < 64)
        __builtin_prefetch(&W[(long long)(colBase + tid) * ldw + (k0 + 32)], 0, 3);
    }
    __syncthreads();

    // Load fragments
    v16bf aH[2], aL[2], bH[2], bL[2];
    #pragma unroll
    for (int t = 0; t < 2; ++t) {
      const int m = wm * 32 + t * 16 + l15;
      #pragma unroll
      for (int j = 0; j < 8; ++j) {
        aH[t][j]     = AsH[m][kA + j];
        aH[t][j + 8] = AsH[m][kB + j];
        aL[t][j]     = AsL[m][kA + j];
        aL[t][j + 8] = AsL[m][kB + j];
      }
    }
    #pragma unroll
    for (int s = 0; s < 2; ++s) {
      const int n = wn * 32 + s * 16 + l15;
      #pragma unroll
      for (int j = 0; j < 8; ++j) {
        bH[s][j]     = WsH[n][kA + j];
        bH[s][j + 8] = WsH[n][kB + j];
        bL[s][j]     = WsL[n][kA + j];
        bL[s][j + 8] = WsL[n][kB + j];
      }
    }

    // 12 WMMAs: hi*hi + hi*lo + lo*hi, all f32-accumulated
    #pragma unroll
    for (int t = 0; t < 2; ++t) {
      #pragma unroll
      for (int s = 0; s < 2; ++s) {
        acc[t][s] = __builtin_amdgcn_wmma_f32_16x16x32_bf16(
            false, aH[t], false, bH[s], (short)0, acc[t][s], false, false);
        acc[t][s] = __builtin_amdgcn_wmma_f32_16x16x32_bf16(
            false, aH[t], false, bL[s], (short)0, acc[t][s], false, false);
        acc[t][s] = __builtin_amdgcn_wmma_f32_16x16x32_bf16(
            false, aL[t], false, bH[s], (short)0, acc[t][s], false, false);
      }
    }
    __syncthreads();
  }

  // Epilogue. C/D layout: VGPR r -> M = r (lanes 0-15) / r+8 (lanes 16-31),
  // N = lane&15.
  #pragma unroll
  for (int t = 0; t < 2; ++t) {
    #pragma unroll
    for (int s = 0; s < 2; ++s) {
      const int col = colBase + wn * 32 + s * 16 + l15;
      const float bv = bias ? bias[col] : 0.0f;
      #pragma unroll
      for (int r = 0; r < 8; ++r) {
        const int row = rowBase + wm * 32 + t * 16 + half * 8 + r;
        float v = acc[t][s][r] + bv;
        if (act == 1) v = (v > 0.0f) ? v : 0.01f * v;
        C[(long long)row * ldc + col] = v;
      }
    }
  }
}

// ---------------------------------------------------------------------------
// Kernel 3: GRU gate fusion -> h_new
// ---------------------------------------------------------------------------
__global__ void gru_kernel(const float* __restrict__ gx,
                           const float* __restrict__ gh,
                           const float* __restrict__ h0,
                           float* __restrict__ h_new) {
  int idx = blockIdx.x * blockDim.x + threadIdx.x;  // over B*H
  if (idx >= B_SZ * H_SZ) return;
  int b = idx >> 9;          // /512
  int h = idx & (H_SZ - 1);
  const float* gxr = gx + (long long)b * 3 * H_SZ;
  const float* ghr = gh + (long long)b * 3 * H_SZ;
  float r = 1.0f / (1.0f + expf(-(gxr[h] + ghr[h])));
  float z = 1.0f / (1.0f + expf(-(gxr[H_SZ + h] + ghr[H_SZ + h])));
  float n = tanhf(gxr[2 * H_SZ + h] + r * ghr[2 * H_SZ + h]);
  h_new[idx] = (1.0f - z) * n + z * h0[idx];
}

// ---------------------------------------------------------------------------
// Kernel 4: in-place log_softmax over rows of V=1024, block 256
// ---------------------------------------------------------------------------
__global__ __launch_bounds__(256)
void log_softmax_kernel(float* __restrict__ logits) {
  __shared__ float red[256];
  const int b = blockIdx.x;
  const int t = threadIdx.x;
  float* row = logits + (long long)b * V_SZ;
  float v0 = row[t], v1 = row[t + 256], v2 = row[t + 512], v3 = row[t + 768];
  float m = fmaxf(fmaxf(v0, v1), fmaxf(v2, v3));
  red[t] = m; __syncthreads();
  for (int s = 128; s > 0; s >>= 1) {
    if (t < s) red[t] = fmaxf(red[t], red[t + s]);
    __syncthreads();
  }
  m = red[0]; __syncthreads();
  float e = expf(v0 - m) + expf(v1 - m) + expf(v2 - m) + expf(v3 - m);
  red[t] = e; __syncthreads();
  for (int s = 128; s > 0; s >>= 1) {
    if (t < s) red[t] += red[t + s];
    __syncthreads();
  }
  const float lse = m + logf(red[0]);
  row[t] = v0 - lse; row[t + 256] = v1 - lse;
  row[t + 512] = v2 - lse; row[t + 768] = v3 - lse;
}

// ---------------------------------------------------------------------------
// Kernel 5: ho_pos_out = sigmoid(ph @ wp2.T + bp2), N=2 (tiny)
// ---------------------------------------------------------------------------
__global__ void hopos_kernel(const float* __restrict__ ph,
                             const float* __restrict__ wp2,
                             const float* __restrict__ bp2,
                             float* __restrict__ out) {
  int t = blockIdx.x * blockDim.x + threadIdx.x;  // over B*2
  if (t >= B_SZ * 2) return;
  int b = t >> 1, j = t & 1;
  const float* pr = ph + (long long)b * H_SZ;
  const float* wr = wp2 + j * H_SZ;
  float s = 0.0f;
  for (int k = 0; k < H_SZ; ++k) s += pr[k] * wr[k];
  s += bp2[j];
  out[t] = 1.0f / (1.0f + expf(-s));
}

// ---------------------------------------------------------------------------
extern "C" void kernel_launch(void* const* d_in, const int* in_sizes, int n_in,
                              void* d_out, int out_size, void* d_ws, size_t ws_size,
                              hipStream_t stream) {
  const float* cond       = (const float*)d_in[0];
  const float* ho_pos     = (const float*)d_in[1];
  const float* hidden     = (const float*)d_in[2];   // [1,B,H] contiguous
  const float* emb_table  = (const float*)d_in[3];
  const float* pos_table  = (const float*)d_in[4];
  const float* w_hp       = (const float*)d_in[5];
  const float* b_hp       = (const float*)d_in[6];
  const float* w_ih       = (const float*)d_in[7];
  const float* w_hh       = (const float*)d_in[8];
  const float* b_ih       = (const float*)d_in[9];
  const float* b_hh       = (const float*)d_in[10];
  const float* wl1        = (const float*)d_in[11];
  const float* bl1        = (const float*)d_in[12];
  const float* wl2        = (const float*)d_in[13];
  const float* bl2        = (const float*)d_in[14];
  const float* wp1        = (const float*)d_in[15];
  const float* bp1        = (const float*)d_in[16];
  const float* wp2        = (const float*)d_in[17];
  const float* bp2        = (const float*)d_in[18];
  const int*   type_label = (const int*)d_in[19];
  const int*   pos        = (const int*)d_in[20];

  // Workspace layout (floats). Needs B*(1282+1536+1536)*4 ~ 272 MB.
  float* ws = (float*)d_ws;
  float* x  = ws;                               // [B, 1282]
  float* gx = x  + (size_t)B_SZ * IN_SZ;        // [B, 1536]
  float* gh = gx + (size_t)B_SZ * 3 * H_SZ;     // [B, 1536]
  float* lh = x;                                // reuse x region: [B, 512]
  float* ph = x + (size_t)B_SZ * H_SZ;          // reuse x region: [B, 512]

  // Output layout (return order): logits[B*V], ho_pos_out[B*2], h_new[B*H]
  float* out_logits = (float*)d_out;
  float* out_hopos  = out_logits + (size_t)B_SZ * V_SZ;
  float* out_h      = out_hopos  + (size_t)B_SZ * 2;

  const dim3 blk(256);

  // 1) assemble x
  {
    long long total = (long long)B_SZ * IN_SZ;
    dim3 grid((unsigned)((total + 255) / 256));
    assemble_x_kernel<<<grid, blk, 0, stream>>>(cond, ho_pos, emb_table,
                                                pos_table, w_hp, b_hp,
                                                type_label, pos, x);
  }
  // 2) gx = x @ w_ih.T + b_ih      [B,1536], K=1282
  gemm_bias_kernel<<<dim3(3 * H_SZ / 64, B_SZ / 128), blk, 0, stream>>>(
      x, IN_SZ, w_ih, IN_SZ, b_ih, gx, 3 * H_SZ, IN_SZ, 0);
  // 3) gh = h0 @ w_hh.T + b_hh     [B,1536], K=512
  gemm_bias_kernel<<<dim3(3 * H_SZ / 64, B_SZ / 128), blk, 0, stream>>>(
      hidden, H_SZ, w_hh, H_SZ, b_hh, gh, 3 * H_SZ, H_SZ, 0);
  // 4) GRU fusion -> h_new (third output, also GEMM input below)
  gru_kernel<<<dim3(B_SZ * H_SZ / 256), blk, 0, stream>>>(gx, gh, hidden, out_h);
  // 5) lh = leaky_relu(h_new @ wl1.T + bl1)   [B,512]
  gemm_bias_kernel<<<dim3(H_SZ / 64, B_SZ / 128), blk, 0, stream>>>(
      out_h, H_SZ, wl1, H_SZ, bl1, lh, H_SZ, H_SZ, 1);
  // 6) logits = lh @ wl2.T + bl2              [B,1024] -> d_out
  gemm_bias_kernel<<<dim3(V_SZ / 64, B_SZ / 128), blk, 0, stream>>>(
      lh, H_SZ, wl2, H_SZ, bl2, out_logits, V_SZ, H_SZ, 0);
  // 7) log_softmax in place
  log_softmax_kernel<<<dim3(B_SZ), blk, 0, stream>>>(out_logits);
  // 8) ph = leaky_relu(h_new @ wp1.T + bp1)   [B,512]
  gemm_bias_kernel<<<dim3(H_SZ / 64, B_SZ / 128), blk, 0, stream>>>(
      out_h, H_SZ, wp1, H_SZ, bp1, ph, H_SZ, H_SZ, 1);
  // 9) ho_pos_out = sigmoid(ph @ wp2.T + bp2) [B,2]
  hopos_kernel<<<dim3(B_SZ * 2 / 256), blk, 0, stream>>>(ph, wp2, bp2, out_hopos);
}
